// HierarchicalReasoningModel_ACTV1Block_30176440221701
// MI455X (gfx1250) — matrix operand order repaired and checked by reading, not compiled, not run on hardware
//
#include <hip/hip_runtime.h>

// ---------------- problem constants ----------------
#define T_TOK 1024
#define HDIM  1024
#define NHEAD 16
#define HEADD 64
#define SEQ   512
#define NEXP  32
#define NDEV  8
#define TOPK  6
#define TOPKD 3
#define INTERN 1536
#define EPSV  1e-5f

typedef __attribute__((ext_vector_type(16))) _Float16 v16h;
typedef __attribute__((ext_vector_type(8)))  _Float16 v8h;
typedef __attribute__((ext_vector_type(8)))  float    v8f;

union F16Frag { v16h v; v8h h[2]; };

// 16-byte async DMA copy global -> LDS (CDNA5 GLOBAL_LOAD_ASYNC_TO_LDS_B128,
// tracked by ASYNCcnt; vdst = LDS byte address VGPR, vaddr = 64-bit address).
__device__ __forceinline__ void async_cp16(void* ldsDst, const void* gsrc) {
  unsigned l = (unsigned)(size_t)ldsDst;
  asm volatile("global_load_async_to_lds_b128 %0, %1, off"
               :: "v"(l), "v"(gsrc) : "memory");
}
__device__ __forceinline__ void async_wait0() {
  asm volatile("s_wait_asynccnt 0x0" ::: "memory");
}

// ---------------- generic WMMA GEMM ----------------
// C[M,N] = alpha * A[M,K] * B[K,N] (+ C if accumulate)
// MODE 0: plain batched (blockIdx.z with strides)
// MODE 1: expert gather: A rows indirected through tokIdx, compact C at offsets[z]
// MODE 2: expert scatter: compact A at offsets[z], C rows scattered via tokIdx with
//         per-row scale tokW and atomicAdd
#define BMT 128
#define BNT 128
#define BKT 32
#define LDSK 48   // padded LDS row stride in halves (96 bytes, 16B aligned)

template<int MODE>
__global__ __launch_bounds__(256) void gemm_wmma(
    const _Float16* __restrict__ A, const _Float16* __restrict__ B, float* __restrict__ C,
    int M, int N, int Kdim, int lda, int ldb, int ldc,
    long long strideA, long long strideB, long long strideC,
    float alpha, int transB, int accumulate,
    const int* __restrict__ counts, const int* __restrict__ offsets,
    const int* __restrict__ tokIdx, const float* __restrict__ tokW)
{
  // double-buffered tiles for async pipeline
  __shared__ __align__(16) _Float16 As[2 * BMT * LDSK];
  __shared__ __align__(16) _Float16 Bs[2 * BNT * LDSK];

  const int tid  = threadIdx.x;
  const int lane = tid & 31;
  const int wave = tid >> 5;
  const int z    = blockIdx.z;

  const _Float16* Ab;
  const _Float16* Bb = B + strideB * z;
  float* Cb;
  int Meff = M;
  const int*   idx  = nullptr;
  const float* wrow = nullptr;
  if (MODE == 0) {
    Ab = A + strideA * z;
    Cb = C + strideC * z;
  } else {
    const int off = offsets[z];
    Meff = counts[z];
    idx  = tokIdx + off;
    wrow = tokW + off;
    if (MODE == 1) { Ab = A;                            Cb = C + (long long)off * ldc; }
    else           { Ab = A + (long long)off * lda;     Cb = C; }
  }

  const int bm0 = blockIdx.y * BMT;
  const int bn0 = blockIdx.x * BNT;
  if (bm0 >= Meff) return;

  const int wr  = wave >> 1;      // 0..3
  const int wc  = wave & 1;       // 0..1
  const int wm0 = wr * 32;
  const int wn0 = wc * 64;

  v8f acc[2][4];
  v8f vzero = {};
  #pragma unroll
  for (int i = 0; i < 2; ++i)
    #pragma unroll
    for (int j = 0; j < 4; ++j)
      acc[i][j] = vzero;

  const int arow_s = tid >> 1;         // 0..127
  const int acol_s = (tid & 1) * 16;   // 0 or 16

  // ---- staging: A tile is a direct strided copy -> async DMA to LDS ----
  auto stageA = [&](int k0, _Float16* dstBase) {
    const int gm = bm0 + arow_s;
    _Float16* dst = dstBase + arow_s * LDSK + acol_s;
    if (gm < Meff) {
      const long long rowA = (MODE == 1) ? (long long)idx[gm] : (long long)gm;
      const _Float16* src = Ab + rowA * lda + (k0 + acol_s);
      async_cp16(dst, src);
      async_cp16(dst + 8, src + 8);
      __builtin_prefetch(src + BKT, 0, 1);   // warm L2 one K-slab ahead
    } else {
      v8h zf = {};
      *(v8h*)(dst)     = zf;
      *(v8h*)(dst + 8) = zf;
    }
  };
  auto stageB = [&](int k0, _Float16* dstBase) {
    if (transB) {
      // B element (k,n) at Bb[n*ldb + k]: rows already K-contiguous -> async DMA
      const int brow = tid >> 1;
      const int bcol = (tid & 1) * 16;
      const int gn = bn0 + brow;
      _Float16* dst = dstBase + brow * LDSK + bcol;
      if (gn < N) {
        const _Float16* src = Bb + (long long)gn * ldb + (k0 + bcol);
        async_cp16(dst, src);
        async_cp16(dst + 8, src + 8);
        __builtin_prefetch(src + BKT, 0, 1);
      } else {
        v8h zf = {};
        *(v8h*)(dst)     = zf;
        *(v8h*)(dst + 8) = zf;
      }
    } else {
      // B element (k,n) at Bb[k*ldb + n]: coalesced read, transposed ds_store
      const int bk  = tid >> 3;          // 0..31
      const int bnc = (tid & 7) * 16;    // 0..112
      _Float16 tmp[16];
      const long long rowoff = (long long)(k0 + bk) * ldb + bn0 + bnc;
      if (bn0 + bnc + 16 <= N) {
        *(v8h*)(tmp)     = *(const v8h*)(Bb + rowoff);
        *(v8h*)(tmp + 8) = *(const v8h*)(Bb + rowoff + 8);
      } else {
        #pragma unroll
        for (int j = 0; j < 16; ++j)
          tmp[j] = (bn0 + bnc + j < N) ? Bb[rowoff + j] : (_Float16)0.0f;
      }
      #pragma unroll
      for (int j = 0; j < 16; ++j)
        dstBase[(bnc + j) * LDSK + bk] = tmp[j];
    }
  };

  // prologue: fill buffer 0
  int cur = 0;
  stageA(0, As);
  stageB(0, Bs);
  async_wait0();
  __syncthreads();

  for (int k0 = 0; k0 < Kdim; k0 += BKT) {
    const int nxt = cur ^ 1;
    // kick off DMA for the next K-slab into the other buffer
    if (k0 + BKT < Kdim) {
      stageA(k0 + BKT, As + nxt * (BMT * LDSK));
      stageB(k0 + BKT, Bs + nxt * (BNT * LDSK));
    }

    // ---- compute on current buffer ----
    {
      const _Float16* AsC = As + cur * (BMT * LDSK);
      const _Float16* BsC = Bs + cur * (BNT * LDSK);
      const int l15   = lane & 15;
      const int ksel  = (lane >> 4) * 8;   // A: lanes 0-15 K{0..7,16..23}, 16-31 K{8..15,24..31}
      const int kselB = (lane >> 4) * 8;   // B: lanes 0-15 K{0..7,16..23}, 16-31 K{8..15,24..31}
      F16Frag af[2], bf[4];
      #pragma unroll
      for (int mi = 0; mi < 2; ++mi) {
        const _Float16* ap = &AsC[(wm0 + mi * 16 + l15) * LDSK + ksel];
        af[mi].h[0] = *(const v8h*)(ap);
        af[mi].h[1] = *(const v8h*)(ap + 16);
      }
      #pragma unroll
      for (int ni = 0; ni < 4; ++ni) {
        const _Float16* bp = &BsC[(wn0 + ni * 16 + l15) * LDSK + kselB];
        bf[ni].h[0] = *(const v8h*)(bp);
        bf[ni].h[1] = *(const v8h*)(bp + 16);
      }
      #pragma unroll
      for (int mi = 0; mi < 2; ++mi)
        #pragma unroll
        for (int ni = 0; ni < 4; ++ni)
          acc[mi][ni] = __builtin_amdgcn_wmma_f32_16x16x32_f16(
              false, af[mi].v, false, bf[ni].v, (short)0, acc[mi][ni], false, false);
    }

    // this wave's DMA into the next buffer must land before the block flips
    async_wait0();
    __syncthreads();
    cur = nxt;
  }

  // ---- epilogue: D layout VGPR g -> M = g + 8*(lane>=16), N = lane&15 ----
  const int l15  = lane & 15;
  const int rsel = (lane >> 4) * 8;
  #pragma unroll
  for (int mi = 0; mi < 2; ++mi) {
    #pragma unroll
    for (int ni = 0; ni < 4; ++ni) {
      #pragma unroll
      for (int g = 0; g < 8; ++g) {
        const int gm = bm0 + wm0 + mi * 16 + rsel + g;
        const int gn = bn0 + wn0 + ni * 16 + l15;
        if (gm < Meff && gn < N) {
          float v = alpha * acc[mi][ni][g];
          if (MODE == 2) {
            atomicAdd(&Cb[(long long)idx[gm] * ldc + gn], wrow[gm] * v);
          } else {
            long long o = (long long)gm * ldc + gn;
            Cb[o] = accumulate ? (Cb[o] + v) : v;
          }
        }
      }
    }
  }
}

// ---------------- elementwise / helper kernels ----------------
__global__ void cvt_f32_f16(const float* __restrict__ in, _Float16* __restrict__ out, long long n) {
  long long i = (long long)blockIdx.x * blockDim.x + threadIdx.x;
  const long long stride = (long long)gridDim.x * blockDim.x;
  for (; i < n; i += stride) out[i] = (_Float16)in[i];
}

__global__ void zero_u32(unsigned int* __restrict__ p, long long n) {
  long long i = (long long)blockIdx.x * blockDim.x + threadIdx.x;
  const long long stride = (long long)gridDim.x * blockDim.x;
  for (; i < n; i += stride) p[i] = 0u;
}

// qkv (T, 3*H) with cols [3][NH][HD] -> Qh/Kh/Vh each (B,NH,S,HD) f16
__global__ void split_qkv(const float* __restrict__ qkv,
                          _Float16* __restrict__ Qh, _Float16* __restrict__ Kh,
                          _Float16* __restrict__ Vh) {
  const int i = blockIdx.x * 256 + threadIdx.x;   // 2^20 total
  const int d = i & 63;
  const int s = (i >> 6) & 511;
  const int h = (i >> 15) & 15;
  const int b = i >> 19;
  const long long row = (long long)(b * SEQ + s) * (3 * HDIM);
  const int col = h * HEADD + d;
  Qh[i] = (_Float16)qkv[row + col];
  Kh[i] = (_Float16)qkv[row + HDIM + col];
  Vh[i] = (_Float16)qkv[row + 2 * HDIM + col];
}

// row softmax over SEQ=512 columns, emit f16 probabilities
__global__ __launch_bounds__(256) void softmax_rows(const float* __restrict__ in,
                                                    _Float16* __restrict__ outH) {
  __shared__ float red[256];
  const long long row = blockIdx.x;
  const int tid = threadIdx.x;
  const float* p = in + row * SEQ;
  float a = p[tid], b = p[tid + 256];
  red[tid] = fmaxf(a, b);
  __syncthreads();
  for (int s = 128; s > 0; s >>= 1) { if (tid < s) red[tid] = fmaxf(red[tid], red[tid + s]); __syncthreads(); }
  const float m = red[0];
  __syncthreads();
  const float e0 = __expf(a - m), e1 = __expf(b - m);
  red[tid] = e0 + e1;
  __syncthreads();
  for (int s = 128; s > 0; s >>= 1) { if (tid < s) red[tid] += red[tid + s]; __syncthreads(); }
  const float inv = 1.0f / red[0];
  _Float16* o = outH + row * SEQ;
  o[tid]       = (_Float16)(e0 * inv);
  o[tid + 256] = (_Float16)(e1 * inv);
}

// O (B,NH,S,HD) f32 -> (B,S,H) f16
__global__ void permute_o(const float* __restrict__ O, _Float16* __restrict__ O2h) {
  const int i = blockIdx.x * 256 + threadIdx.x;
  const int d = i & 63;
  const int s = (i >> 6) & 511;
  const int h = (i >> 15) & 15;
  const int b = i >> 19;
  O2h[(long long)(b * SEQ + s) * HDIM + h * HEADD + d] = (_Float16)O[i];
}

// out = rms_norm(a + b (+ c)), HDIM cols, one block per row
__global__ __launch_bounds__(256) void resid_rms(const float* __restrict__ a,
                                                 const float* __restrict__ b,
                                                 const float* __restrict__ c,
                                                 float* __restrict__ outF,
                                                 _Float16* __restrict__ outH) {
  __shared__ float red[256];
  const long long row = blockIdx.x;
  const int tid = threadIdx.x;
  float y[4];
  float ss = 0.0f;
  #pragma unroll
  for (int j = 0; j < 4; ++j) {
    const long long o = row * HDIM + tid + j * 256;
    float v = a[o];
    if (b) v += b[o];
    if (c) v += c[o];
    y[j] = v;
    ss += v * v;
  }
  red[tid] = ss;
  __syncthreads();
  for (int s = 128; s > 0; s >>= 1) { if (tid < s) red[tid] += red[tid + s]; __syncthreads(); }
  const float scale = rsqrtf(red[0] / (float)HDIM + EPSV);
  #pragma unroll
  for (int j = 0; j < 4; ++j) {
    const long long o = row * HDIM + tid + j * 256;
    if (outF) outF[o] = y[j] * scale;
    if (outH) outH[o] = (_Float16)(y[j] * scale);
  }
}

// act[r, c] = silu(gu[r, c]) * gu[r, INTER + c], gu rows of width 2*INTER
__global__ void swiglu_k(const float* __restrict__ gu, _Float16* __restrict__ act, long long nrows) {
  long long i = (long long)blockIdx.x * 256 + threadIdx.x;
  const long long total = nrows * INTERN;
  const long long stride = (long long)gridDim.x * 256;
  for (; i < total; i += stride) {
    const long long r = i / INTERN;
    const int cidx = (int)(i - r * INTERN);
    const float g = gu[r * (2 * INTERN) + cidx];
    const float u = gu[r * (2 * INTERN) + INTERN + cidx];
    const float s = g / (1.0f + __expf(-g));
    act[i] = (_Float16)(s * u);
  }
}

// router logits: one wave per token, one lane per expert (E=32 == wave32)
__global__ __launch_bounds__(256) void router_logits(const float* __restrict__ h1,
                                                     const float* __restrict__ Wr,
                                                     float* __restrict__ logits) {
  const int lane = threadIdx.x & 31;
  const int t = blockIdx.x * 8 + (threadIdx.x >> 5);
  if (t >= T_TOK) return;
  const float* x = h1 + (long long)t * HDIM;
  float s = 0.0f;
  for (int k = 0; k < HDIM; ++k) s += x[k] * Wr[k * NEXP + lane];
  logits[t * NEXP + lane] = s;
}

// per-token routing: device top-3, expert top-6, softmax weights, build per-expert lists
__global__ void route_kernel(const float* __restrict__ logits,
                             int* __restrict__ counts,
                             int* __restrict__ etokTmp, float* __restrict__ ewTmp,
                             float* __restrict__ probsSum, float* __restrict__ devHit) {
  const int t = blockIdx.x * blockDim.x + threadIdx.x;
  if (t >= T_TOK) return;
  float p[NEXP];
  float mx = -1e30f;
  for (int e = 0; e < NEXP; ++e) { p[e] = logits[t * NEXP + e]; mx = fmaxf(mx, p[e]); }
  float s = 0.0f;
  for (int e = 0; e < NEXP; ++e) { p[e] = __expf(p[e] - mx); s += p[e]; }
  const float inv = 1.0f / s;
  for (int e = 0; e < NEXP; ++e) p[e] *= inv;

  float ds[NDEV];
  for (int d = 0; d < NDEV; ++d)
    ds[d] = p[4 * d] + p[4 * d + 1] + p[4 * d + 2] + p[4 * d + 3];

  bool dsel[NDEV];
  for (int d = 0; d < NDEV; ++d) dsel[d] = false;
  for (int it = 0; it < TOPKD; ++it) {
    int best = -1; float bv = -1e30f;
    for (int d = 0; d < NDEV; ++d)
      if (!dsel[d] && ds[d] > bv) { bv = ds[d]; best = d; }
    dsel[best] = true;
  }

  int sel[TOPK]; float w[TOPK];
  bool esel[NEXP];
  for (int e = 0; e < NEXP; ++e) esel[e] = false;
  for (int it = 0; it < TOPK; ++it) {
    int best = -1; float bv = -1e30f;
    for (int e = 0; e < NEXP; ++e)
      if (!esel[e] && dsel[e >> 2] && p[e] > bv) { bv = p[e]; best = e; }
    esel[best] = true;
    sel[it] = best;
    w[it] = bv;
  }
  // softmax over selected weights (w[0] is the max; keep it before overwriting)
  const float wmax = w[0];
  float sw = 0.0f;
  for (int it = 0; it < TOPK; ++it) { w[it] = __expf(w[it] - wmax); sw += w[it]; }
  const float iw = 1.0f / sw;

  bool dhit[NDEV];
  for (int d = 0; d < NDEV; ++d) dhit[d] = false;
  for (int it = 0; it < TOPK; ++it) {
    const int e = sel[it];
    const int pos = atomicAdd(&counts[e], 1);
    etokTmp[e * T_TOK + pos] = t;
    ewTmp[e * T_TOK + pos] = w[it] * iw;
    dhit[e >> 2] = true;
  }
  for (int e = 0; e < NEXP; ++e) atomicAdd(&probsSum[e], p[e]);
  for (int d = 0; d < NDEV; ++d) if (dhit[d]) atomicAdd(&devHit[d], 1.0f);
}

__global__ void scan_kernel(const int* __restrict__ counts, int* __restrict__ offsets) {
  if (threadIdx.x == 0 && blockIdx.x == 0) {
    int a = 0;
    for (int e = 0; e < NEXP; ++e) { offsets[e] = a; a += counts[e]; }
    offsets[NEXP] = a;
  }
}

__global__ void compact_kernel(const int* __restrict__ counts, const int* __restrict__ offsets,
                               const int* __restrict__ etokTmp, const float* __restrict__ ewTmp,
                               int* __restrict__ tokIdx, float* __restrict__ tokW) {
  const int i = blockIdx.x * 256 + threadIdx.x;   // 32*1024
  const int e = i >> 10, j = i & 1023;
  if (j < counts[e]) {
    tokIdx[offsets[e] + j] = etokTmp[i];
    tokW[offsets[e] + j]   = ewTmp[i];
  }
}

__global__ void aux_kernel(const int* __restrict__ counts, const float* __restrict__ probsSum,
                           const float* __restrict__ devHit, float* __restrict__ outAux) {
  if (threadIdx.x != 0 || blockIdx.x != 0) return;
  float f[NEXP], P[NEXP];
  for (int e = 0; e < NEXP; ++e) {
    f[e] = (float)counts[e] / ((float)T_TOK * TOPK + 1e-10f);
    P[e] = probsSum[e] / (float)T_TOK;
  }
  float eb = 0.0f;
  for (int e = 0; e < NEXP; ++e) eb += f[e] * P[e];
  eb = fminf(eb * 0.003f, 10.0f);
  float db = 0.0f, cb = 0.0f;
  for (int d = 0; d < NDEV; ++d) {
    const float devf = 0.25f * (f[4 * d] + f[4 * d + 1] + f[4 * d + 2] + f[4 * d + 3]);
    const float devP = P[4 * d] + P[4 * d + 1] + P[4 * d + 2] + P[4 * d + 3];
    db += devf * devP;
    const float fc = devHit[d] / ((float)T_TOK * TOPKD + 1e-10f);
    cb += fc * devP;
  }
  db = fminf(db * 0.05f, 10.0f);
  cb = fminf(cb * 0.02f, 10.0f);
  *outAux = eb + db + cb;
}

// ---------------- host launcher ----------------
extern "C" void kernel_launch(void* const* d_in, const int* in_sizes, int n_in,
                              void* d_out, int out_size, void* d_ws, size_t ws_size,
                              hipStream_t stream) {
  const float* x    = (const float*)d_in[0];
  const float* Wqkv = (const float*)d_in[1];
  const float* Wo   = (const float*)d_in[2];
  const float* WguS = (const float*)d_in[3];
  const float* WdS  = (const float*)d_in[4];
  const float* Wr   = (const float*)d_in[5];
  const float* Wgu  = (const float*)d_in[6];
  const float* Wd   = (const float*)d_in[7];
  float* out = (float*)d_out;

  char* bump = (char*)d_ws;
  auto alloc = [&](size_t bytes) -> void* {
    void* r = (void*)bump;
    bump += (bytes + 255) & ~(size_t)255;
    return r;
  };

  const long long nTH = 1048576;   // T*H
  _Float16* xh     = (_Float16*)alloc(2 * nTH);
  _Float16* Wqkv_h = (_Float16*)alloc(2LL * 1024 * 3072);
  _Float16* Wo_h   = (_Float16*)alloc(2LL * 1024 * 1024);
  _Float16* WguS_h = (_Float16*)alloc(2LL * 2 * 1024 * 3072);
  _Float16* WdS_h  = (_Float16*)alloc(2LL * 2 * 1536 * 1024);
  _Float16* Wgu_h  = (_Float16*)alloc(2LL * 32 * 1024 * 3072);
  _Float16* Wd_h   = (_Float16*)alloc(2LL * 32 * 1536 * 1024);
  float* qkvF      = (float*)alloc(4LL * 1024 * 3072);
  _Float16* Qh     = (_Float16*)alloc(2 * nTH);
  _Float16* Kh     = (_Float16*)alloc(2 * nTH);
  _Float16* Vh     = (_Float16*)alloc(2 * nTH);
  float* scores    = (float*)alloc(4LL * 32 * 512 * 512);
  _Float16* Ph     = (_Float16*)alloc(2LL * 32 * 512 * 512);
  float* Obuf      = (float*)alloc(4 * nTH);
  _Float16* O2h    = (_Float16*)alloc(2 * nTH);
  float* attnO     = (float*)alloc(4 * nTH);
  float* h1        = (float*)alloc(4 * nTH);
  _Float16* h1h    = (_Float16*)alloc(2 * nTH);
  float* guBuf     = (float*)alloc(4LL * 1024 * 3072);
  _Float16* actSh  = (_Float16*)alloc(2LL * 1024 * 1536);
  float* sharedB   = (float*)alloc(4 * nTH);
  float* logits    = (float*)alloc(4LL * 1024 * 32);
  int* countsB     = (int*)alloc(4 * 32);
  int* offsetsB    = (int*)alloc(4 * 33);
  int* etokTmp     = (int*)alloc(4LL * 32 * 1024);
  float* ewTmp     = (float*)alloc(4LL * 32 * 1024);
  int* tokIdx      = (int*)alloc(4 * 6144);
  float* tokW      = (float*)alloc(4 * 6144);
  float* probsSum  = (float*)alloc(4 * 32);
  float* devHit    = (float*)alloc(4 * 8);
  float* GUc       = (float*)alloc(4LL * 6144 * 3072);
  _Float16* actC   = (_Float16*)alloc(2LL * 6144 * 1536);
  float* rBuf      = (float*)alloc(4 * nTH);

  // --- weight / activation conversion to f16 ---
  cvt_f32_f16<<<2048, 256, 0, stream>>>(x, xh, nTH);
  cvt_f32_f16<<<4096, 256, 0, stream>>>(Wqkv, Wqkv_h, 3145728LL);
  cvt_f32_f16<<<2048, 256, 0, stream>>>(Wo, Wo_h, 1048576LL);
  cvt_f32_f16<<<4096, 256, 0, stream>>>(WguS, WguS_h, 6291456LL);
  cvt_f32_f16<<<4096, 256, 0, stream>>>(WdS, WdS_h, 3145728LL);
  cvt_f32_f16<<<8192, 256, 0, stream>>>(Wgu, Wgu_h, 100663296LL);
  cvt_f32_f16<<<8192, 256, 0, stream>>>(Wd, Wd_h, 50331648LL);

  // --- zero accumulators / routing state ---
  zero_u32<<<1024, 256, 0, stream>>>((unsigned int*)rBuf, nTH);
  zero_u32<<<1, 64, 0, stream>>>((unsigned int*)countsB, 32);
  zero_u32<<<1, 64, 0, stream>>>((unsigned int*)probsSum, 32);
  zero_u32<<<1, 32, 0, stream>>>((unsigned int*)devHit, 8);

  // --- QKV projection: (1024,1024) x (1024,3072) ---
  gemm_wmma<0><<<dim3(24, 8, 1), 256, 0, stream>>>(
      xh, Wqkv_h, qkvF, 1024, 3072, 1024, 1024, 3072, 3072,
      0LL, 0LL, 0LL, 1.0f, 0, 0, nullptr, nullptr, nullptr, nullptr);
  split_qkv<<<4096, 256, 0, stream>>>(qkvF, Qh, Kh, Vh);

  // --- scores = Q K^T / 8, batched over 32 (b,h) ---
  gemm_wmma<0><<<dim3(4, 4, 32), 256, 0, stream>>>(
      Qh, Kh, scores, 512, 512, 64, 64, 64, 512,
      (long long)512 * 64, (long long)512 * 64, (long long)512 * 512,
      0.125f, 1, 0, nullptr, nullptr, nullptr, nullptr);
  softmax_rows<<<16384, 256, 0, stream>>>(scores, Ph);

  // --- O = P V, batched ---
  gemm_wmma<0><<<dim3(1, 4, 32), 256, 0, stream>>>(
      Ph, Vh, Obuf, 512, 64, 512, 512, 64, 64,
      (long long)512 * 512, (long long)512 * 64, (long long)512 * 64,
      1.0f, 0, 0, nullptr, nullptr, nullptr, nullptr);
  permute_o<<<4096, 256, 0, stream>>>(Obuf, O2h);

  // --- attention output projection ---
  gemm_wmma<0><<<dim3(8, 8, 1), 256, 0, stream>>>(
      O2h, Wo_h, attnO, 1024, 1024, 1024, 1024, 1024, 1024,
      0LL, 0LL, 0LL, 1.0f, 0, 0, nullptr, nullptr, nullptr, nullptr);

  // --- h1 = rms_norm(x + attn) ---
  resid_rms<<<1024, 256, 0, stream>>>(x, attnO, nullptr, h1, h1h);

  // --- shared experts: 2x SwiGLU accumulated into sharedB ---
  for (int e = 0; e < 2; ++e) {
    gemm_wmma<0><<<dim3(24, 8, 1), 256, 0, stream>>>(
        h1h, WguS_h + (long long)e * 1024 * 3072, guBuf, 1024, 3072, 1024,
        1024, 3072, 3072, 0LL, 0LL, 0LL, 1.0f, 0, 0,
        nullptr, nullptr, nullptr, nullptr);
    swiglu_k<<<4096, 256, 0, stream>>>(guBuf, actSh, 1024LL);
    gemm_wmma<0><<<dim3(8, 8, 1), 256, 0, stream>>>(
        actSh, WdS_h + (long long)e * 1536 * 1024, sharedB, 1024, 1024, 1536,
        1536, 1024, 1024, 0LL, 0LL, 0LL, 1.0f, 0, e,
        nullptr, nullptr, nullptr, nullptr);
  }

  // --- routing ---
  router_logits<<<128, 256, 0, stream>>>(h1, Wr, logits);
  route_kernel<<<4, 256, 0, stream>>>(logits, countsB, etokTmp, ewTmp, probsSum, devHit);
  scan_kernel<<<1, 1, 0, stream>>>(countsB, offsetsB);
  compact_kernel<<<128, 256, 0, stream>>>(countsB, offsetsB, etokTmp, ewTmp, tokIdx, tokW);

  // --- routed experts: gather GEMM -> swiglu -> scatter GEMM ---
  gemm_wmma<1><<<dim3(24, 8, 32), 256, 0, stream>>>(
      h1h, Wgu_h, GUc, 1024, 3072, 1024, 1024, 3072, 3072,
      0LL, (long long)1024 * 3072, 0LL, 1.0f, 0, 0,
      countsB, offsetsB, tokIdx, tokW);
  swiglu_k<<<8192, 256, 0, stream>>>(GUc, actC, 6144LL);   // total rows = T*K always
  gemm_wmma<2><<<dim3(8, 8, 32), 256, 0, stream>>>(
      actC, Wd_h, rBuf, 1024, 1024, 1536, 1536, 1024, 1024,
      0LL, (long long)1536 * 1024, 0LL, 1.0f, 0, 0,
      countsB, offsetsB, tokIdx, tokW);

  // --- final: out = rms_norm(h1 + shared + r), aux scalar ---
  resid_rms<<<1024, 256, 0, stream>>>(h1, sharedB, rBuf, out, nullptr);
  aux_kernel<<<1, 1, 0, stream>>>(countsB, probsSum, devHit, out + nTH);
}
